// C2QAttention_6665789243605
// MI455X (gfx1250) — matrix operand order repaired and checked by reading, not compiled, not run on hardware
//
#include <hip/hip_runtime.h>
#include <hip/hip_bf16.h>
#include <cstdint>

typedef __bf16 bf16;
typedef __attribute__((ext_vector_type(16))) __bf16 v16bf;
typedef __attribute__((ext_vector_type(8)))  __bf16 v8bf;
typedef __attribute__((ext_vector_type(8)))  float   v8f;
typedef __attribute__((ext_vector_type(4)))  unsigned u32x4;
typedef __attribute__((ext_vector_type(8)))  unsigned u32x8;

#define B_   32
#define TC_  512
#define TQ_  512
#define H_   1024

// LDS layout of the attention kernel (dynamic smem, base offset 0):
#define SP_BYTES   (128 * 512 * 2)          // 131072: score/prob tile
#define SA_OFF     (SP_BYTES)               // 131072: A staging 128x32 bf16
#define SB_OFF     (SP_BYTES + 8192)        // 139264: B staging 128x32 bf16

// ---------- fragment helpers (wave32, v_wmma_f32_16x16x32_bf16 layouts) ----------

// A fragment (16x32 bf16), LDS tile row-major [m][k] with row stride `ld` halves.
__device__ __forceinline__ v16bf load_a_frag(const bf16* lds, int m0, int k0, int ld) {
  unsigned lane = threadIdx.x & 31u;
  const bf16* p = lds + (size_t)(m0 + (int)(lane & 15u)) * ld + k0 + (int)((lane >> 4) << 3);
  v8bf lo = *(const v8bf*)(p);
  v8bf hi = *(const v8bf*)(p + 16);
  v16bf r;
#pragma unroll
  for (int i = 0; i < 8; ++i) { r[i] = lo[i]; r[i + 8] = hi[i]; }
  return r;
}

// B fragment (32x16 bf16), LDS tile stored as [n][k] (row stride 32 halves), K-chunk = 32.
__device__ __forceinline__ v16bf load_b_frag(const bf16* lds, int n0) {
  unsigned lane = threadIdx.x & 31u;
  const bf16* p = lds + (size_t)(n0 + (int)(lane & 15u)) * 32 + (int)((lane >> 4) << 3);
  v8bf lo = *(const v8bf*)(p);
  v8bf hi = *(const v8bf*)(p + 16);
  v16bf r;
#pragma unroll
  for (int i = 0; i < 8; ++i) { r[i] = lo[i]; r[i + 8] = hi[i]; }
  return r;
}

__device__ __forceinline__ v8f wmma_bf16(v16bf a, v16bf b, v8f c) {
  return __builtin_amdgcn_wmma_f32_16x16x32_bf16(false, a, false, b, (short)0, c, false, false);
}

__device__ __forceinline__ v8f zero_v8f() {
  v8f z;
#pragma unroll
  for (int i = 0; i < 8; ++i) z[i] = 0.0f;
  return z;
}

// ---------- Tensor Data Mover: 2D bf16 tile (rows x 32 halves) global -> LDS ----------
// Builds D# group0/group1 per CDNA5 ISA (CDNA5 ISA) and issues
// tensor_load_to_lds via inline asm (portable across builtin-arity differences).
// LDS receives the tile packed row-major: row j at lds_addr + j*64B == [n][k=32] layout.
__device__ __forceinline__ void tdm_load_tile_bf16(const bf16* gsrc, unsigned lds_byte_off,
                                                   unsigned rows) {
  unsigned long long ga = (unsigned long long)(uintptr_t)gsrc;
  u32x4 g0;
  g0[0] = 1u;                                        // count=1, user mode, no gather
  g0[1] = lds_byte_off;                              // lds_addr
  g0[2] = (unsigned)(ga & 0xFFFFFFFFu);              // global_addr[31:0]
  g0[3] = (unsigned)((ga >> 32) & 0x1FFFFFFu)        // global_addr[56:32]
          | 0x80000000u;                             // type = 2 ("image")
  u32x8 g1;
  g1[0] = (1u << 16);                                // data_size=1 (2 bytes); mask/pad/iter = 0
  g1[1] = ((unsigned)H_ & 0xFFFFu) << 16;            // tensor_dim0[15:0]
  g1[2] = ((unsigned)H_ >> 16)                       // tensor_dim0[31:16]
          | (((unsigned)TQ_ & 0xFFFFu) << 16);       // tensor_dim1[15:0]
  g1[3] = ((unsigned)TQ_ >> 16) | (32u << 16);       // tensor_dim1[31:16] | tile_dim0=32
  g1[4] = rows;                                      // tile_dim1 | tile_dim2=0
  g1[5] = (unsigned)H_;                              // tensor_dim0_stride[31:0] = row pitch
  g1[6] = 0u;                                        // stride hi | dim1_stride lo
  g1[7] = 0u;                                        // dim1_stride hi
  asm volatile("tensor_load_to_lds %0, %1" :: "s"(g0), "s"(g1) : "memory");
}

// ---------- kernel 1: o_qt = o_q @ W^T + b   (bf16 out to workspace) ----------
// Flattened GEMM: M = B*Tq = 16384, N = H = 1024, K = H = 1024.
__global__ __launch_bounds__(256)
void linear_bf16_kernel(const float* __restrict__ o_q, const float* __restrict__ Wm,
                        const float* __restrict__ bias, bf16* __restrict__ oqt) {
  extern __shared__ char smem[];
  bf16* sA = (bf16*)smem;            // 128 x 32 bf16 (o_q tile, [m][k])
  bf16* sB = (bf16*)(smem + 8192);   // 128 x 32 bf16 (W rows,  [n][k])

  const int M0 = blockIdx.x * 128;
  const int N0 = blockIdx.y * 128;
  const unsigned t = threadIdx.x;
  const unsigned w = t >> 5, lane = t & 31u;
  const int wm = (int)(w & 3u) * 32;
  const int wn = (int)(w >> 2) * 64;
  const int hi8 = (int)((lane >> 4) << 3);
  const int nlane = (int)(lane & 15u);
  const unsigned lr = t >> 1;
  const unsigned lc = (t & 1u) << 4;

  v8f acc[2][4];
#pragma unroll
  for (int r = 0; r < 2; ++r)
#pragma unroll
    for (int c = 0; c < 4; ++c) acc[r][c] = zero_v8f();

  for (int kk = 0; kk < H_; kk += 32) {
    { // stage A: o_q f32 -> bf16
      const float* g = o_q + (size_t)(M0 + (int)lr) * H_ + kk + (int)lc;
      __builtin_prefetch(g + 32, 0, 1);
      float tmp[16];
#pragma unroll
      for (int i = 0; i < 4; ++i) *(float4*)(tmp + 4 * i) = ((const float4*)g)[i];
      bf16* d = sA + lr * 32 + lc;
#pragma unroll
      for (int i = 0; i < 16; ++i) d[i] = (bf16)tmp[i];
    }
    { // stage B: W rows f32 -> bf16 ([n][k] == what B-frag wants)
      const float* g = Wm + (size_t)(N0 + (int)lr) * H_ + kk + (int)lc;
      float tmp[16];
#pragma unroll
      for (int i = 0; i < 4; ++i) *(float4*)(tmp + 4 * i) = ((const float4*)g)[i];
      bf16* d = sB + lr * 32 + lc;
#pragma unroll
      for (int i = 0; i < 16; ++i) d[i] = (bf16)tmp[i];
    }
    __syncthreads();
    v16bf af[2], bfr[4];
    af[0] = load_a_frag(sA, wm,      0, 32);
    af[1] = load_a_frag(sA, wm + 16, 0, 32);
#pragma unroll
    for (int c = 0; c < 4; ++c) bfr[c] = load_b_frag(sB, wn + c * 16);
#pragma unroll
    for (int r = 0; r < 2; ++r)
#pragma unroll
      for (int c = 0; c < 4; ++c) acc[r][c] = wmma_bf16(af[r], bfr[c], acc[r][c]);
    __syncthreads();
  }

  // epilogue: + bias, cvt bf16, store (oqt is re-read -> keep cacheable)
#pragma unroll
  for (int c = 0; c < 4; ++c) {
    const int col = N0 + wn + c * 16 + nlane;
    const float bv = bias[col];
#pragma unroll
    for (int r = 0; r < 2; ++r)
#pragma unroll
      for (int i = 0; i < 8; ++i) {
        const int row = M0 + wm + r * 16 + i + hi8;
        oqt[(size_t)row * H_ + col] = (bf16)(acc[r][c][i] + bv);
      }
  }
}

// ---------- kernel 2: scores -> softmax -> context, one (batch, 128-row) tile ----------
__global__ __launch_bounds__(256)
void c2q_attn_kernel(const float* __restrict__ o_c, const float* __restrict__ o_q,
                     const bf16* __restrict__ oqt, const int* __restrict__ q_len,
                     const int* __restrict__ c_len, float* __restrict__ out) {
  extern __shared__ char smem[];
  bf16* sP = (bf16*)smem;               // 128 x 512 (scores, then probs)
  bf16* sA = (bf16*)(smem + SA_OFF);    // 128 x 32
  bf16* sB = (bf16*)(smem + SB_OFF);    // 128 x 32

  const int bb = blockIdx.y;
  const int i0 = blockIdx.x * 128;
  const unsigned t = threadIdx.x;
  const unsigned w = t >> 5, lane = t & 31u;
  const int wm = (int)(w & 3u) * 32;
  const int wn = (int)(w >> 2) * 64;
  const int hi8 = (int)((lane >> 4) << 3);
  const int nlane = (int)(lane & 15u);
  const unsigned lr = t >> 1;
  const unsigned lc = (t & 1u) << 4;
  const float scale = 0.03125f;  // 1/sqrt(1024)

  // ---- Phase 1: S = (o_c @ oqt^T) * scale  -> sP as bf16 ----
  for (int np = 0; np < 4; ++np) {
    const int N0 = np * 128;  // j range
    v8f acc[2][4];
#pragma unroll
    for (int r = 0; r < 2; ++r)
#pragma unroll
      for (int c = 0; c < 4; ++c) acc[r][c] = zero_v8f();

    for (int kk = 0; kk < H_; kk += 32) {
      // B tile via TDM: oqt rows [j = N0..N0+127][k = kk..kk+31] (bf16, layout-preserving DMA)
      if (w == 0) {
        tdm_load_tile_bf16(oqt + ((size_t)(bb * TQ_ + N0)) * H_ + kk, SB_OFF, 128u);
      }
      { // A tile (all threads, overlaps the TDM): o_c f32 -> bf16
        const float* g = o_c + ((size_t)(bb * TC_ + i0 + (int)lr)) * H_ + kk + (int)lc;
        __builtin_prefetch(g + 32, 0, 1);
        float tmp[16];
#pragma unroll
        for (int i = 0; i < 4; ++i) *(float4*)(tmp + 4 * i) = ((const float4*)g)[i];
        bf16* d = sA + lr * 32 + lc;
#pragma unroll
        for (int i = 0; i < 16; ++i) d[i] = (bf16)tmp[i];
      }
      if (w == 0) __builtin_amdgcn_s_wait_tensorcnt(0);
      __syncthreads();
      v16bf af[2], bfr[4];
      af[0] = load_a_frag(sA, wm,      0, 32);
      af[1] = load_a_frag(sA, wm + 16, 0, 32);
#pragma unroll
      for (int c = 0; c < 4; ++c) bfr[c] = load_b_frag(sB, wn + c * 16);
#pragma unroll
      for (int r = 0; r < 2; ++r)
#pragma unroll
        for (int c = 0; c < 4; ++c) acc[r][c] = wmma_bf16(af[r], bfr[c], acc[r][c]);
      __syncthreads();
    }
#pragma unroll
    for (int c = 0; c < 4; ++c)
#pragma unroll
      for (int r = 0; r < 2; ++r)
#pragma unroll
        for (int i = 0; i < 8; ++i) {
          const int row = wm + r * 16 + i + hi8;
          const int col = N0 + wn + c * 16 + nlane;
          sP[(size_t)row * TQ_ + col] = (bf16)(acc[r][c][i] * scale);
        }
  }
  __syncthreads();

  // ---- Phase 2: masked softmax over j (one thread per row) ----
  if (t < 128) {
    const int gi = i0 + (int)t;
    bf16* rowp = sP + (size_t)t * TQ_;
    const int cl = c_len[bb];
    if (gi >= cl) {
      v8bf z;
#pragma unroll
      for (int i = 0; i < 8; ++i) z[i] = (bf16)0.0f;
      for (int j = 0; j < TQ_; j += 8) *(v8bf*)(rowp + j) = z;
    } else {
      int qn = q_len[bb];
      if (qn > TQ_) qn = TQ_;
      float m = -3.0e38f;
      for (int j = 0; j < qn; ++j) m = fmaxf(m, (float)rowp[j]);
      float s = 0.0f;
      for (int j = 0; j < qn; ++j) s += __expf((float)rowp[j] - m);
      const float inv = 1.0f / s;
      for (int j = 0; j < qn; ++j) rowp[j] = (bf16)(__expf((float)rowp[j] - m) * inv);
      for (int j = qn; j < TQ_; ++j) rowp[j] = (bf16)0.0f;  // q-mask => prob 0
    }
  }
  __syncthreads();

  // ---- Phase 3: context = P @ o_q  (A frags straight from sP) ----
  const unsigned tj = t >> 3;          // 0..31  (j within chunk)
  const unsigned th = (t & 7u) << 4;   // 0,16,..112 (h base)
  for (int hp = 0; hp < 8; ++hp) {
    const int N0 = hp * 128;  // h range
    v8f acc[2][4];
#pragma unroll
    for (int r = 0; r < 2; ++r)
#pragma unroll
      for (int c = 0; c < 4; ++c) acc[r][c] = zero_v8f();

    for (int jj = 0; jj < TQ_; jj += 32) {
      { // B: o_q[j][h] f32 -> transposed bf16 LDS [h][j]
        const float* g = o_q + ((size_t)(bb * TQ_ + jj + (int)tj)) * H_ + N0 + (int)th;
        __builtin_prefetch(g + 32 * H_, 0, 1);
        float tmp[16];
#pragma unroll
        for (int i = 0; i < 4; ++i) *(float4*)(tmp + 4 * i) = ((const float4*)g)[i];
#pragma unroll
        for (int i = 0; i < 16; ++i) sB[(size_t)((int)th + i) * 32 + tj] = (bf16)tmp[i];
      }
      __syncthreads();
      v16bf af[2], bfr[4];
      af[0] = load_a_frag(sP, wm,      jj, TQ_);
      af[1] = load_a_frag(sP, wm + 16, jj, TQ_);
#pragma unroll
      for (int c = 0; c < 4; ++c) bfr[c] = load_b_frag(sB, wn + c * 16);
#pragma unroll
      for (int r = 0; r < 2; ++r)
#pragma unroll
        for (int c = 0; c < 4; ++c) acc[r][c] = wmma_bf16(af[r], bfr[c], acc[r][c]);
      __syncthreads();
    }
    // epilogue: f32 out, nontemporal (never re-read; keep L2 for oqt/W)
#pragma unroll
    for (int c = 0; c < 4; ++c)
#pragma unroll
      for (int r = 0; r < 2; ++r)
#pragma unroll
        for (int i = 0; i < 8; ++i) {
          const int grow = i0 + wm + r * 16 + i + hi8;
          const int col  = N0 + wn + c * 16 + nlane;
          __builtin_nontemporal_store(acc[r][c][i],
                                      &out[((size_t)(bb * TC_) + grow) * H_ + col]);
        }
  }
}

// ---------- host entry ----------
extern "C" void kernel_launch(void* const* d_in, const int* in_sizes, int n_in,
                              void* d_out, int out_size, void* d_ws, size_t ws_size,
                              hipStream_t stream) {
  (void)in_sizes; (void)n_in; (void)out_size; (void)ws_size;
  const float* o_c  = (const float*)d_in[0];
  const float* o_q  = (const float*)d_in[1];
  const float* Wm   = (const float*)d_in[2];
  const float* bias = (const float*)d_in[3];
  const int* q_len  = (const int*)d_in[4];
  const int* c_len  = (const int*)d_in[5];
  float* out = (float*)d_out;
  bf16* oqt = (bf16*)d_ws;  // B*Tq*H bf16 = 32 MB scratch

  dim3 g1((B_ * TQ_) / 128, H_ / 128);  // 128 x 8 blocks
  linear_bf16_kernel<<<g1, 256, 16384, stream>>>(o_q, Wm, bias, oqt);

  dim3 g2(TC_ / 128, B_);               // 4 x 32 blocks
  size_t smem2 = (size_t)SP_BYTES + 2 * 8192;  // 147456 B
  c2q_attn_kernel<<<g2, 256, smem2, stream>>>(o_c, o_q, oqt, q_len, c_len, out);
}
